// LogicGatedSNN_103079215204
// MI455X (gfx1250) — matrix operand order repaired and checked by reading, not compiled, not run on hardware
//
#include <hip/hip_runtime.h>

// LogicGatedSNN step for MI455X (gfx1250, wave32).
//
// out layout (float32, concatenated in reference return order):
//   [0 .. 8192)                     spikes
//   [8192 .. 16384)                 new_v_mem
//   [16384 .. 16384+8192*8192)      new_trace
//   [16384+8192*8192 .. +8192)      new_vth

#define IN_F   8192
#define OUT_F  8192
#define THRESH 50.0f

typedef __attribute__((ext_vector_type(16))) _Float16 v16h;
typedef __attribute__((ext_vector_type(8)))  float    v8f;
typedef __attribute__((ext_vector_type(4)))  float    f4;

// Matches the pointee type expected by the async-to-LDS builtin:
// "__attribute__((__vector_size__(4 * sizeof(int)))) int"
typedef int b128_t __attribute__((vector_size(16)));
typedef __attribute__((address_space(1))) b128_t* gptr_b128;
typedef __attribute__((address_space(3))) b128_t* lptr_b128;

#if defined(__has_builtin)
#if __has_builtin(__builtin_amdgcn_global_load_async_to_lds_b128)
#define HAVE_ASYNC_LDS 1
#endif
#endif

__device__ __forceinline__ _Float16 binarize(float s) {
    return s > THRESH ? (_Float16)1.0f : (_Float16)0.0f;
}

// One wave (32 lanes) per block; each wave computes 16 output rows of the
// binarized GEMV via v_wmma_f32_16x16x32_f16, then does the neuron update.
__global__ __launch_bounds__(32) void snn_gemv_wmma(
    const float* __restrict__ x,        // [IN_F] spike input (0/1)
    const float* __restrict__ states,   // [OUT_F, IN_F]
    const float* __restrict__ mp,       // [OUT_F]
    const float* __restrict__ vth,      // [OUT_F]
    float* __restrict__ out_spikes,     // [OUT_F]
    float* __restrict__ out_v,          // [OUT_F]
    float* __restrict__ out_vth)        // [OUT_F]
{
    __shared__ alignas(32) _Float16 xs[IN_F];

    const int lane    = threadIdx.x;            // 0..31, full wave
    const int rowBase = blockIdx.x * 16;

#if HAVE_ASYNC_LDS
    // Stage raw f32 x into LDS with async DMA (ASYNCcnt-tracked), then
    // convert LDS f32 -> LDS f16 once.
    __shared__ alignas(16) float xraw[IN_F];
    for (int i = lane * 4; i < IN_F; i += 32 * 4) {
        __builtin_amdgcn_global_load_async_to_lds_b128(
            (gptr_b128)(x + i),
            (lptr_b128)(xraw + i),
            /*offset=*/0, /*cpol=*/0);
    }
    asm volatile("s_wait_asynccnt 0" ::: "memory");
    for (int i = lane * 4; i < IN_F; i += 32 * 4) {
        f4 v = *(const f4*)(xraw + i);
        xs[i + 0] = (_Float16)v[0];
        xs[i + 1] = (_Float16)v[1];
        xs[i + 2] = (_Float16)v[2];
        xs[i + 3] = (_Float16)v[3];
    }
#else
    // Stage x into LDS as f16 (0/1 values are exact). float4 global loads,
    // 8-byte LDS stores.
    for (int i = lane * 4; i < IN_F; i += 32 * 4) {
        f4 v = *(const f4*)(x + i);
        xs[i + 0] = (_Float16)v[0];
        xs[i + 1] = (_Float16)v[1];
        xs[i + 2] = (_Float16)v[2];
        xs[i + 3] = (_Float16)v[3];
    }
#endif
    __syncthreads();

    // A-matrix lane mapping (16-bit A 16x32):
    //   row   = lane & 15
    //   halves 0..7  -> K = kb..kb+7        (kb = (lane>>4)*8)
    //   halves 8..15 -> K = kb+16..kb+23
    const int mrow  = lane & 15;
    const int khalf = lane >> 4;                // 0 or 1
    const float* srow = states + (size_t)(rowBase + mrow) * IN_F;

    v8f c = {};
    for (int kt = 0; kt < IN_F; kt += 32) {
        const int kb = kt + khalf * 8;

        // Prefetch the stream ~16 K-steps ahead into L2.
        __builtin_prefetch(srow + kb + 512, /*rw=*/0, /*locality=*/0);

        // 16 states elements per lane as four b128 loads; single-use stream,
        // so non-temporal to keep L2 for reused data.
        f4 s0 = __builtin_nontemporal_load((const f4*)(srow + kb));
        f4 s1 = __builtin_nontemporal_load((const f4*)(srow + kb + 4));
        f4 s2 = __builtin_nontemporal_load((const f4*)(srow + kb + 16));
        f4 s3 = __builtin_nontemporal_load((const f4*)(srow + kb + 20));

        v16h a;
#pragma unroll
        for (int j = 0; j < 4; ++j) {
            a[j]      = binarize(s0[j]);
            a[4 + j]  = binarize(s1[j]);
            a[8 + j]  = binarize(s2[j]);
            a[12 + j] = binarize(s3[j]);
        }

        // B-matrix (16-bit B 32x16): same K order as A: elements 0..7 ->
        // K = kh..kh+7, elements 8..15 -> K = kh+16..kh+23 (kh = khalf*8);
        // x broadcast across all 16 N columns.
        v16h b;
#pragma unroll
        for (int j = 0; j < 8; ++j) {
            b[j]     = xs[kt + khalf * 8 + j];
            b[8 + j] = xs[kt + 16 + khalf * 8 + j];
        }

        c = __builtin_amdgcn_wmma_f32_16x16x32_f16(
                /*neg_a=*/false, a, /*neg_b=*/false, b,
                /*c_mod=*/(short)0, c, /*reuse_a=*/false, /*reuse_b=*/false);
    }

    // D (f32 16x16): lane 0 VGPR r = row r, lane 16 VGPR r = row 8+r.
    // Columns are identical (B broadcast), so lanes 0 and 16 finish 8 rows each.
    if ((lane & 15) == 0) {
        const int base = rowBase + khalf * 8;
#pragma unroll
        for (int r = 0; r < 8; ++r) {
            const int row  = base + r;
            const float cur = c[r];
            const float v   = mp[row] * 0.8f + cur;
            const float th  = vth[row];
            const float sp  = (v >= th) ? 1.0f : 0.0f;
            out_spikes[row] = sp;
            out_v[row]      = v * (1.0f - sp) * 0.2f;
            float nth = th + (sp - 0.05f) * 0.1f;
            nth = fminf(fmaxf(nth, 0.5f), 10.0f);
            out_vth[row] = nth;
        }
    }
}

// Elementwise trace update: one block per output row, float4 streaming.
// new_trace = clip(trace*0.9 + spikes[o]*x[i], 0, 5)
// trace in/out are pure single-pass streams -> non-temporal; x is re-read by
// all 8192 blocks -> default (L2-resident) hints.
__global__ __launch_bounds__(256) void snn_trace(
    const float* __restrict__ x,          // [IN_F]
    const float* __restrict__ trace,      // [OUT_F, IN_F]
    const float* __restrict__ spikes,     // [OUT_F] (written by snn_gemv_wmma)
    float* __restrict__ out_trace)        // [OUT_F, IN_F]
{
    const int row  = blockIdx.x;
    const float s  = spikes[row];
    const size_t rbase = (size_t)row * (IN_F / 4);
    const f4* tr = (const f4*)trace + rbase;
    const f4* xv = (const f4*)x;
    f4*       ot = (f4*)out_trace + rbase;

    for (int i = threadIdx.x; i < IN_F / 4; i += 256) {
        f4 t  = __builtin_nontemporal_load(tr + i);
        f4 xx = xv[i];
        f4 o;
#pragma unroll
        for (int j = 0; j < 4; ++j) {
            float v = t[j] * 0.9f + s * xx[j];
            o[j] = fminf(fmaxf(v, 0.0f), 5.0f);
        }
        __builtin_nontemporal_store(o, ot + i);
    }
}

extern "C" void kernel_launch(void* const* d_in, const int* in_sizes, int n_in,
                              void* d_out, int out_size, void* d_ws, size_t ws_size,
                              hipStream_t stream) {
    const float* x      = (const float*)d_in[0];  // spike_input        [8192]
    const float* states = (const float*)d_in[1];  // synapse_states     [8192*8192]
    const float* mp     = (const float*)d_in[2];  // membrane_potential [8192]
    const float* vth    = (const float*)d_in[3];  // adaptive_threshold [8192]
    const float* trace  = (const float*)d_in[4];  // eligibility_trace  [8192*8192]

    float* out        = (float*)d_out;
    float* out_spikes = out;                                    // 8192
    float* out_v      = out + OUT_F;                            // 8192
    float* out_trace  = out + 2 * OUT_F;                        // 8192*8192
    float* out_vth    = out + 2 * OUT_F + (size_t)OUT_F * IN_F; // 8192

    // Phase 1: binarized GEMV (WMMA) + neuron update. 512 waves, 16 rows each.
    snn_gemv_wmma<<<OUT_F / 16, 32, 0, stream>>>(
        x, states, mp, vth, out_spikes, out_v, out_vth);

    // Phase 2: trace stream (dominant 512 MB of traffic) at max parallelism.
    snn_trace<<<OUT_F, 256, 0, stream>>>(x, trace, out_spikes, out_trace);
}
